// GCN_21818433863980
// MI455X (gfx1250) — matrix operand order repaired and checked by reading, not compiled, not run on hardware
//
#include <hip/hip_runtime.h>

typedef __bf16 bf16;
typedef __attribute__((ext_vector_type(16))) __bf16 v16bf;
typedef __attribute__((ext_vector_type(8)))  __bf16 v8bf;
typedef __attribute__((ext_vector_type(8)))  float  v8f;

// ======================================================================
// Fragment-layout packing (per CDNA5 ISA 16-bit WMMA layouts).
// ======================================================================

__global__ void k_pack_a(const float* __restrict__ X, bf16* __restrict__ Ap,
                         long nthreads, int K) {
  long idx = (long)blockIdx.x * blockDim.x + threadIdx.x;
  if (idx >= nthreads) return;
  int half = (int)(idx & 1);
  int l    = (int)((idx >> 1) & 31);
  long tile = idx >> 6;
  int KT = K >> 5;
  long tm = tile / KT;
  int  kt = (int)(tile % KT);
  int m  = (int)(tm * 16) + (l & 15);
  int k  = kt * 32 + ((l & 16) ? 8 : 0) + half * 16;
  const float* src = X + (size_t)m * K + k;
  float4 f0 = *(const float4*)(src);
  float4 f1 = *(const float4*)(src + 4);
  v8bf o;
  o[0] = (bf16)f0.x; o[1] = (bf16)f0.y; o[2] = (bf16)f0.z; o[3] = (bf16)f0.w;
  o[4] = (bf16)f1.x; o[5] = (bf16)f1.y; o[6] = (bf16)f1.z; o[7] = (bf16)f1.w;
  *(v8bf*)(Ap + (tile * 32 + l) * 16 + half * 8) = o;
}

__global__ void k_pack_a_bias_relu(const float* __restrict__ Y, const float* __restrict__ b,
                                   bf16* __restrict__ Ap, long nthreads, int K) {
  long idx = (long)blockIdx.x * blockDim.x + threadIdx.x;
  if (idx >= nthreads) return;
  int half = (int)(idx & 1);
  int l    = (int)((idx >> 1) & 31);
  long tile = idx >> 6;
  int KT = K >> 5;
  long tm = tile / KT;
  int  kt = (int)(tile % KT);
  int m  = (int)(tm * 16) + (l & 15);
  int k  = kt * 32 + ((l & 16) ? 8 : 0) + half * 16;
  const float* src = Y + (size_t)m * K + k;
  float4 f0 = *(const float4*)(src);
  float4 f1 = *(const float4*)(src + 4);
  float4 b0 = *(const float4*)(b + k);
  float4 b1 = *(const float4*)(b + k + 4);
  v8bf o;
  o[0] = (bf16)fmaxf(f0.x + b0.x, 0.f); o[1] = (bf16)fmaxf(f0.y + b0.y, 0.f);
  o[2] = (bf16)fmaxf(f0.z + b0.z, 0.f); o[3] = (bf16)fmaxf(f0.w + b0.w, 0.f);
  o[4] = (bf16)fmaxf(f1.x + b1.x, 0.f); o[5] = (bf16)fmaxf(f1.y + b1.y, 0.f);
  o[6] = (bf16)fmaxf(f1.z + b1.z, 0.f); o[7] = (bf16)fmaxf(f1.w + b1.w, 0.f);
  *(v8bf*)(Ap + (tile * 32 + l) * 16 + half * 8) = o;
}

__global__ void k_pack_b(const float* __restrict__ W, bf16* __restrict__ Bp,
                         long nthreads, int K, int N) {
  long idx = (long)blockIdx.x * blockDim.x + threadIdx.x;
  if (idx >= nthreads) return;
  int l = (int)(idx & 31);
  long tile = idx >> 5;
  int KT = K >> 5;
  int tn = (int)(tile / KT);
  int kt = (int)(tile % KT);
  int n  = tn * 16 + (l & 15);
  int kb = kt * 32 + ((l & 16) ? 8 : 0);
  v16bf o;
#pragma unroll
  for (int e = 0; e < 16; ++e) o[e] = (bf16)W[(size_t)(kb + e + (e & 8)) * N + n];
  ((v16bf*)Bp)[idx] = o;
}

// ======================================================================
// WMMA GEMM, B staged in LDS (8 waves / block share one n-strip).
// ======================================================================
template <int NT>
__global__ void k_gemm_lds(const bf16* __restrict__ Ap, const bf16* __restrict__ Bp,
                           float* __restrict__ C, int M, int K, int N) {
  extern __shared__ char smem[];
  bf16* Bl = (bf16*)smem;
  int KT  = K >> 5;
  int ntg = N / (16 * NT);
  int tid = threadIdx.x;
  int lane = tid & 31;
  int wv   = tid >> 5;
  int wpb  = blockDim.x >> 5;
  int mtiles = M >> 4;

  int tg  = blockIdx.x % ntg;
  int tm  = (blockIdx.x / ntg) * wpb + wv;
  if (tm >= mtiles) tm = mtiles - 1;            // duplicate tile: benign identical writes

  {
    const uint4* src = (const uint4*)(Bp + (size_t)(tg * NT) * KT * 512);
    uint4* dst = (uint4*)Bl;
    int n16 = NT * KT * 64;
    for (int i = tid; i < n16; i += blockDim.x) dst[i] = src[i];
  }
  __syncthreads();

  v8f acc[NT] = {};
  const v16bf* ap = (const v16bf*)Ap + ((size_t)tm * KT) * 32 + lane;
  const v16bf* bl = (const v16bf*)Bl + lane;
  for (int kt = 0; kt < KT; ++kt) {
    __builtin_prefetch(ap + (size_t)(kt + 4) * 32, 0, 1);
    v16bf av = ap[(size_t)kt * 32];
#pragma unroll
    for (int t = 0; t < NT; ++t) {
      v16bf bv = bl[(size_t)(t * KT + kt) * 32];
      acc[t] = __builtin_amdgcn_wmma_f32_16x16x32_bf16(false, av, false, bv,
                                                       (short)0, acc[t], false, false);
    }
  }
  int crow  = tm * 16 + ((lane & 16) ? 8 : 0);
  int nbase = tg * NT * 16 + (lane & 15);
#pragma unroll
  for (int t = 0; t < NT; ++t) {
    float* cp = C + (size_t)crow * N + nbase + t * 16;
#pragma unroll
    for (int r = 0; r < 8; ++r) cp[(size_t)r * N] = acc[t][r];
  }
}

// ======================================================================
// CSR build: histogram -> exclusive scan -> fill.
// ======================================================================
__global__ void k_zero_u4(uint4* __restrict__ p, long n16) {
  long i = (long)blockIdx.x * blockDim.x + threadIdx.x;
  if (i >= n16) return;
  p[i] = make_uint4(0u, 0u, 0u, 0u);
}

__global__ void k_hist(const int* __restrict__ rows, int* __restrict__ cnt, long E) {
  long i = (long)blockIdx.x * blockDim.x + threadIdx.x;
  if (i >= E) return;
  atomicAdd(&cnt[rows[i]], 1);
}

// single block, 1024 threads: exclusive scan of cnt[0..n) -> offs[0..n]; cursor = offs.
// cnt may alias cursor.
__global__ void k_scan(const int* __restrict__ cnt, int* __restrict__ offs,
                       int* __restrict__ cursor, int n) {
  __shared__ int sums[1024];
  int tid = threadIdx.x;
  int per = (n + (int)blockDim.x - 1) / (int)blockDim.x;
  int start = tid * per;
  int end = min(start + per, n);
  int s = 0;
  for (int i = start; i < end; ++i) s += cnt[i];
  sums[tid] = s;
  __syncthreads();
  for (int d = 1; d < 1024; d <<= 1) {
    int v = (tid >= d) ? sums[tid - d] : 0;
    __syncthreads();
    sums[tid] += v;
    __syncthreads();
  }
  int base = (tid == 0) ? 0 : sums[tid - 1];
  for (int i = start; i < end; ++i) {
    int c = cnt[i];
    offs[i] = base;
    cursor[i] = base;
    base += c;
  }
  if (end == n) offs[n] = base;
}

__global__ void k_fill(const int* __restrict__ rows, const int* __restrict__ cols,
                       const float* __restrict__ vals, int* __restrict__ cursor,
                       int* __restrict__ ocol, float* __restrict__ oval, long E) {
  long i = (long)blockIdx.x * blockDim.x + threadIdx.x;
  if (i >= E) return;
  int r = rows[i];
  int pos = atomicAdd(&cursor[r], 1);
  ocol[pos] = cols[i];
  oval[pos] = vals[i];
}

// ======================================================================
// Row-parallel CSR SpMM: wave per row, register accumulation, no atomics.
// ======================================================================
template <int FPL>
__global__ void k_spmm_csr(const int* __restrict__ offs, const int* __restrict__ cols,
                           const float* __restrict__ vals, const float* __restrict__ H,
                           float* __restrict__ Y, int nrows, int F) {
  int row = (int)((((long)blockIdx.x * blockDim.x) + threadIdx.x) >> 5);
  int lane = threadIdx.x & 31;
  if (row >= nrows) return;
  int s = offs[row], e = offs[row + 1];
  float acc[FPL];
#pragma unroll
  for (int j = 0; j < FPL; ++j) acc[j] = 0.f;
  const float* hb = H + lane * FPL;
  for (int i = s; i < e; ++i) {
    int c = cols[i];
    float v = vals[i];
    const float* hp = hb + (size_t)c * F;
    if constexpr (FPL == 8) {
      float4 h0 = *(const float4*)(hp);
      float4 h1 = *(const float4*)(hp + 4);
      acc[0] += v * h0.x; acc[1] += v * h0.y; acc[2] += v * h0.z; acc[3] += v * h0.w;
      acc[4] += v * h1.x; acc[5] += v * h1.y; acc[6] += v * h1.z; acc[7] += v * h1.w;
    } else {
      float2 h2 = *(const float2*)(hp);
      acc[0] += v * h2.x; acc[1] += v * h2.y;
    }
  }
  float* yp = Y + (size_t)row * F + lane * FPL;
  if constexpr (FPL == 8) {
    *(float4*)(yp)     = make_float4(acc[0], acc[1], acc[2], acc[3]);
    *(float4*)(yp + 4) = make_float4(acc[4], acc[5], acc[6], acc[7]);
  } else {
    *(float2*)(yp) = make_float2(acc[0], acc[1]);
  }
}

// ---------------- out = log_softmax(A + b) over 64 classes; wave per row ----------------
__global__ void k_bias_logsoftmax64(const float* __restrict__ A, const float* __restrict__ b,
                                    float* __restrict__ out, long M) {
  long wid = (((long)blockIdx.x * blockDim.x) + threadIdx.x) >> 5;
  int lane = threadIdx.x & 31;
  if (wid >= M) return;
  float2 t  = *(const float2*)(A + wid * 64 + lane * 2);
  float2 bb = *(const float2*)(b + lane * 2);
  float x0 = t.x + bb.x, x1 = t.y + bb.y;
  float m = fmaxf(x0, x1);
#pragma unroll
  for (int off = 16; off > 0; off >>= 1) m = fmaxf(m, __shfl_xor(m, off, 32));
  float e = __expf(x0 - m) + __expf(x1 - m);
#pragma unroll
  for (int off = 16; off > 0; off >>= 1) e += __shfl_xor(e, off, 32);
  float lse = m + __logf(e);
  float2 o; o.x = x0 - lse; o.y = x1 - lse;
  *(float2*)(out + wid * 64 + lane * 2) = o;
}

extern "C" void kernel_launch(void* const* d_in, const int* in_sizes, int n_in,
                              void* d_out, int out_size, void* d_ws, size_t ws_size,
                              hipStream_t stream) {
  const float* x    = (const float*)d_in[0];
  const int*   erow = (const int*)d_in[1];
  const int*   ecol = (const int*)d_in[2];
  const float* eval = (const float*)d_in[3];
  const float* W1   = (const float*)d_in[4];
  const float* b1   = (const float*)d_in[5];
  const float* W2   = (const float*)d_in[6];
  const float* b2   = (const float*)d_in[7];

  int  nhid   = in_sizes[5];                 // 256
  int  nclass = in_sizes[7];                 // 64
  int  nfeat  = in_sizes[4] / nhid;          // 512
  long M      = (long)in_sizes[0] / nfeat;   // 100000
  long E      = in_sizes[1];                 // 3200000

  char* w = (char*)d_ws;
  size_t off = 0;
  auto take = [&](size_t bytes) -> void* {
    void* p = w + off;
    off += (bytes + 255) & ~(size_t)255;
    return p;
  };
  bf16*  W1p   = (bf16*)take((size_t)nfeat * nhid * 2);
  bf16*  W2p   = (bf16*)take((size_t)nhid * nclass * 2);
  bf16*  Xp    = (bf16*)take((size_t)M * nfeat * 2);
  float* H0    = (float*)take((size_t)M * nhid * 4);
  float* Y1    = (float*)take((size_t)M * nhid * 4);
  bf16*  Hp    = (bf16*)take((size_t)M * nhid * 2);
  float* H2    = (float*)take((size_t)M * nclass * 4);
  float* A2    = (float*)take((size_t)M * nclass * 4);
  int*   offs  = (int*)take((size_t)(M + 1) * 4);
  int*   curs  = (int*)take((size_t)M * 4);       // also the histogram counts
  int*   csrC  = (int*)take((size_t)E * 4);
  float* csrV  = (float*)take((size_t)E * 4);

  const int BT = 256; // 8 waves per block
  auto cdiv = [](long a, long b) { return (int)((a + b - 1) / b); };

  // ---- CSR build (graph reused by both layers) ----
  k_zero_u4<<<cdiv(M * 4 / 16, BT), BT, 0, stream>>>((uint4*)curs, M * 4 / 16);
  k_hist<<<cdiv(E, BT), BT, 0, stream>>>(erow, curs, E);
  k_scan<<<1, 1024, 0, stream>>>(curs, offs, curs, (int)M);
  k_fill<<<cdiv(E, BT), BT, 0, stream>>>(erow, ecol, eval, curs, csrC, csrV, E);

  // ---- pack weights and X into WMMA fragment layouts ----
  long nbt1 = (long)(nfeat / 32) * (nhid / 16) * 32;
  long nbt2 = (long)(nhid / 32) * (nclass / 16) * 32;
  k_pack_b<<<cdiv(nbt1, BT), BT, 0, stream>>>(W1, W1p, nbt1, nfeat, nhid);
  k_pack_b<<<cdiv(nbt2, BT), BT, 0, stream>>>(W2, W2p, nbt2, nhid, nclass);
  long nat1 = (M / 16) * (nfeat / 32) * 64;
  k_pack_a<<<cdiv(nat1, BT), BT, 0, stream>>>(x, Xp, nat1, nfeat);

  // ---- layer 1 GEMM: H0 = X @ W1 ----
  {
    int mtiles = (int)(M / 16), ntg = nhid / 64, KT = nfeat / 32;
    int blocks = cdiv(mtiles, 8) * ntg;
    size_t lds = (size_t)4 * KT * 1024;      // 64 KB
    k_gemm_lds<4><<<blocks, BT, lds, stream>>>(Xp, W1p, H0, (int)M, nfeat, nhid);
  }

  // ---- layer 1 SpMM (row-parallel, no atomics): Y1 = A_sp @ H0 ----
  k_spmm_csr<8><<<cdiv(M * 32, BT), BT, 0, stream>>>(offs, csrC, csrV, H0, Y1, (int)M, nhid);

  // ---- h = relu(Y1 + b1), packed into layer-2 A fragments ----
  long nat2 = (M / 16) * (nhid / 32) * 64;
  k_pack_a_bias_relu<<<cdiv(nat2, BT), BT, 0, stream>>>(Y1, b1, Hp, nat2, nhid);

  // ---- layer 2 GEMM: H2 = h @ W2 ----
  {
    int mtiles = (int)(M / 16), ntg = nclass / 64, KT = nhid / 32;
    int blocks = cdiv(mtiles, 8) * ntg;
    size_t lds = (size_t)4 * KT * 1024;      // 32 KB
    k_gemm_lds<4><<<blocks, BT, lds, stream>>>(Hp, W2p, H2, (int)M, nhid, nclass);
  }

  // ---- layer 2 SpMM ----
  k_spmm_csr<2><<<cdiv(M * 32, BT), BT, 0, stream>>>(offs, csrC, csrV, H2, A2, (int)M, nclass);

  // ---- out = log_softmax(A2 + b2) ----
  k_bias_logsoftmax64<<<cdiv(M * 32, BT), BT, 0, stream>>>(A2, b2, (float*)d_out, M);
}
